// SlidingWindowAttention3d_9165460210373
// MI455X (gfx1250) — matrix operand order repaired and checked by reading, not compiled, not run on hardware
//
#include <hip/hip_runtime.h>
#include <hip/hip_bf16.h>

// ---------------- problem constants (from reference) ----------------
#define BB      4
#define DD      8
#define HH      56
#define WW      56
#define NN      (DD*HH*WW)        // 25088
#define CC      128
#define NHEADS  8
#define HDIM    16
#define MTOT    (BB*NN)           // 100352 rows
#define LOCAL   27

typedef __attribute__((ext_vector_type(16))) _Float16 v16h;
typedef __attribute__((ext_vector_type(8)))  _Float16 v8h;
typedef __attribute__((ext_vector_type(4)))  _Float16 v4h;
typedef __attribute__((ext_vector_type(8)))  float    v8f;

__device__ __forceinline__ v16h cat16(v8h lo, v8h hi) {
    return __builtin_shufflevector(lo, hi, 0,1,2,3,4,5,6,7,8,9,10,11,12,13,14,15);
}

// Async global->LDS copy of 16 bytes per lane (CDNA5 GLOBAL_LOAD_ASYNC_TO_LDS_B128)
__device__ __forceinline__ void async_copy_b128(const _Float16* gsrc, _Float16* ldst) {
    const unsigned lds_off = (unsigned)(unsigned long long)ldst;   // flat->LDS truncation
    const unsigned long long ga = (unsigned long long)gsrc;
    asm volatile("global_load_async_to_lds_b128 %0, %1, off"
                 :: "v"(lds_off), "v"(ga) : "memory");
}
__device__ __forceinline__ void wait_async0() {
    asm volatile("s_wait_asynccnt 0x0" ::: "memory");
}

// =====================================================================
// Kernel 0: convert q_w||kv_w||proj_w (f32) to fused f16 weight buffer
// layout: [0,384)x128 = qkv channels, [384,512)x128 = proj
// =====================================================================
__global__ __launch_bounds__(256)
void wconv_kernel(const float* __restrict__ q_w, const float* __restrict__ kv_w,
                  const float* __restrict__ pw, _Float16* __restrict__ wf16)
{
    const int idx  = blockIdx.x * 256 + threadIdx.x;   // 16384 threads, 4 elems each
    const int base = idx * 4;
    const float* src;
    if      (base < 128 * CC) src = q_w + base;
    else if (base < 384 * CC) src = kv_w + (base - 128 * CC);
    else                      src = pw + (base - 384 * CC);
    const float4 f = *(const float4*)src;
    v4h h;
    h[0] = (_Float16)f.x; h[1] = (_Float16)f.y;
    h[2] = (_Float16)f.z; h[3] = (_Float16)f.w;
    *(v4h*)(wf16 + base) = h;
}

// =====================================================================
// Kernel 1: fused QKV GEMM  y[M,384] = x[M,128] @ Wqkv^T + bias
// epilogue: l2norm per head (q,k), +query_embedding (q), store f16.
// block = 256 threads (8 waves), tile 64x64, K loop 4x32.
// =====================================================================
__global__ __launch_bounds__(256)
void qkv_gemm_kernel(const float* __restrict__ x,
                     const _Float16* __restrict__ wqkv,
                     const float* __restrict__ q_b, const float* __restrict__ kv_b,
                     const float* __restrict__ qe,
                     _Float16* __restrict__ qws, _Float16* __restrict__ kws,
                     _Float16* __restrict__ vws)
{
    __shared__ _Float16 As[64 * 40];   // 64 rows x 32 K, stride 40 halves (80B)
    __shared__ _Float16 Bs[64 * 40];

    const int tid  = threadIdx.x;
    const int lane = tid & 31;
    const int wave = tid >> 5;
    const int rowBase = blockIdx.x * 64;
    const int colBase = blockIdx.y * 64;
    const int subM  = wave >> 1;        // 0..3
    const int subN0 = (wave & 1) * 2;   // 0 or 2

    // per-thread staging coordinates
    const int ar = tid >> 3, ac = (tid & 7) * 4;     // A: float4 chunk (2 per thread)
    const int br = tid >> 2, bc = (tid & 3) * 8;     // B: one 16B chunk per thread

    v8f acc0 = {};
    v8f acc1 = {};

    for (int kk = 0; kk < 4; ++kk) {
        const int k0 = kk * 32;
        // B tile: async global->LDS, one b128 per lane (4096 B total)
        async_copy_b128(wqkv + (size_t)(colBase + br) * CC + k0 + bc,
                        &Bs[br * 40 + bc]);
        // A tile: f32 -> f16, vectorized (float4 load, 8B LDS store)
        #pragma unroll
        for (int half = 0; half < 2; ++half) {
            const int r = ar + half * 32;
            const float4 f = *(const float4*)&x[(size_t)(rowBase + r) * CC + k0 + ac];
            v4h h;
            h[0] = (_Float16)f.x; h[1] = (_Float16)f.y;
            h[2] = (_Float16)f.z; h[3] = (_Float16)f.w;
            *(v4h*)&As[r * 40 + ac] = h;
        }
        if (kk < 3)
            __builtin_prefetch(x + (size_t)(rowBase + ar) * CC + k0 + 32 + ac, 0, 3);
        wait_async0();
        __syncthreads();

        // A fragment: rows m=lane&15; halves 0..7 -> K ka..ka+7, 8..15 -> ka+16..+23
        const int mrow = subM * 16 + (lane & 15);
        const int ka   = ((lane >> 4) & 1) * 8;
        v16h a = cat16(*(const v8h*)&As[mrow * 40 + ka],
                       *(const v8h*)&As[mrow * 40 + ka + 16]);

        // B fragment: col n=lane&15; halves 0..7 -> K kb..kb+7, 8..15 -> kb+16..+23
        const int kb    = ((lane >> 4) & 1) * 8;
        const int nrow0 = subN0 * 16 + (lane & 15);
        v16h bf0 = cat16(*(const v8h*)&Bs[nrow0 * 40 + kb],
                         *(const v8h*)&Bs[nrow0 * 40 + kb + 16]);
        v16h bf1 = cat16(*(const v8h*)&Bs[(nrow0 + 16) * 40 + kb],
                         *(const v8h*)&Bs[(nrow0 + 16) * 40 + kb + 16]);

        acc0 = __builtin_amdgcn_wmma_f32_16x16x32_f16(false, a, false, bf0,
                                                      (short)0, acc0, false, false);
        acc1 = __builtin_amdgcn_wmma_f32_16x16x32_f16(false, a, false, bf1,
                                                      (short)0, acc1, false, false);
        __syncthreads();
    }

    // Epilogue. C/D layout: lane l, reg r -> row m = r + 8*(l>=16), col n = l&15.
    const int n  = lane & 15;
    const int hs = (lane >> 4) & 1;
    union { v8f v; float f[8]; } u0, u1;
    u0.v = acc0; u1.v = acc1;

    #pragma unroll
    for (int t = 0; t < 2; ++t) {
        const float* af = t ? u1.f : u0.f;
        const int gc  = colBase + (subN0 + t) * 16 + n;
        const int seg = gc >> 7;           // 0=q 1=k 2=v (uniform per wave-tile)
        const int lc  = gc & 127;
        const float bias = (seg == 0) ? q_b[gc] : kv_b[gc - CC];
        const float qev  = (seg == 0) ? qe[gc] : 0.f;
        #pragma unroll
        for (int r = 0; r < 8; ++r) {
            float d = af[r] + bias;
            const int gm = rowBase + subM * 16 + hs * 8 + r;
            if (seg < 2) {  // l2norm over the 16 head channels (16 lanes)
                float ss = d * d;
                ss += __shfl_xor(ss, 1, 16);
                ss += __shfl_xor(ss, 2, 16);
                ss += __shfl_xor(ss, 4, 16);
                ss += __shfl_xor(ss, 8, 16);
                d = d / fmaxf(sqrtf(ss), 1e-12f);
                if (seg == 0) d += qev;
            }
            const _Float16 hv = (_Float16)d;
            const size_t off = (size_t)gm * CC + lc;
            if      (seg == 0) qws[off] = hv;
            else if (seg == 1) kws[off] = hv;
            else               vws[off] = hv;
        }
    }
}

// =====================================================================
// Kernel 2: 27-neighbor sliding-window attention, one thread per (b,h,n)
// =====================================================================
__global__ __launch_bounds__(256)
void swin_attn_kernel(const _Float16* __restrict__ qws,
                      const _Float16* __restrict__ kws,
                      const _Float16* __restrict__ vws,
                      _Float16* __restrict__ ows,
                      const float* __restrict__ temperature,
                      const float* __restrict__ rpb)
{
    const int idx  = blockIdx.x * 256 + threadIdx.x;   // grid covers B*h*N exactly
    const int n    = idx % NN;
    const int bh   = idx / NN;
    const int head = bh & (NHEADS - 1);
    const int b    = bh >> 3;

    const int z   = n / (HH * WW);
    const int rem = n - z * (HH * WW);
    const int y   = rem / WW;
    const int xx  = rem - y * WW;
    const size_t rowb = (size_t)b * NN;

    float qv[16];
    {
        const _Float16* qp = qws + (rowb + n) * CC + head * HDIM;
        v8h lo = *(const v8h*)qp;
        v8h hi = *(const v8h*)(qp + 8);
        #pragma unroll
        for (int i = 0; i < 8; ++i) { qv[i] = (float)lo[i]; qv[8 + i] = (float)hi[i]; }
    }

    float logits[LOCAL];
    unsigned vmask = 0;
    int cnt = 0;
    #pragma unroll
    for (int j = 0; j < LOCAL; ++j) {
        const int dz = j / 9 - 1, dy = (j / 3) % 3 - 1, dx = j % 3 - 1;
        const int zz = z + dz, yy = y + dy, xn = xx + dx;
        const bool ok = ((unsigned)zz < (unsigned)DD) &
                        ((unsigned)yy < (unsigned)HH) &
                        ((unsigned)xn < (unsigned)WW);
        float dot = 0.f;
        if (ok) {
            const int nn = (zz * HH + yy) * WW + xn;
            const _Float16* kp = kws + (rowb + nn) * CC + head * HDIM;
            v8h lo = *(const v8h*)kp;
            v8h hi = *(const v8h*)(kp + 8);
            #pragma unroll
            for (int i = 0; i < 8; ++i) {
                dot = fmaf(qv[i],     (float)lo[i], dot);
                dot = fmaf(qv[8 + i], (float)hi[i], dot);
            }
            vmask |= 1u << j;
            ++cnt;
        }
        logits[j] = dot;
    }

    // softplus(temperature[h]) * log(valid_count)  (SEQ_SCALE recomputed on device)
    const float sc = log1pf(__expf(temperature[head])) * logf((float)cnt);

    float mx = -1e30f;
    #pragma unroll
    for (int j = 0; j < LOCAL; ++j)
        if ((vmask >> j) & 1u) {
            logits[j] = logits[j] * sc + rpb[head * LOCAL + j];
            mx = fmaxf(mx, logits[j]);
        }
    float sum = 0.f;
    #pragma unroll
    for (int j = 0; j < LOCAL; ++j)
        if ((vmask >> j) & 1u) {
            const float e = __expf(logits[j] - mx);
            logits[j] = e;
            sum += e;
        }
    const float inv = 1.f / sum;

    float acc[16] = {};
    #pragma unroll
    for (int j = 0; j < LOCAL; ++j)
        if ((vmask >> j) & 1u) {
            const float w = logits[j] * inv;
            const int dz = j / 9 - 1, dy = (j / 3) % 3 - 1, dx = j % 3 - 1;
            const int nn = ((z + dz) * HH + (y + dy)) * WW + (xx + dx);
            const _Float16* vp = vws + (rowb + nn) * CC + head * HDIM;
            v8h lo = *(const v8h*)vp;
            v8h hi = *(const v8h*)(vp + 8);
            #pragma unroll
            for (int i = 0; i < 8; ++i) {
                acc[i]     = fmaf(w, (float)lo[i], acc[i]);
                acc[8 + i] = fmaf(w, (float)hi[i], acc[8 + i]);
            }
        }

    v8h o0, o1;
    #pragma unroll
    for (int i = 0; i < 8; ++i) { o0[i] = (_Float16)acc[i]; o1[i] = (_Float16)acc[8 + i]; }
    _Float16* op = ows + (rowb + n) * CC + head * HDIM;
    *(v8h*)op       = o0;
    *(v8h*)(op + 8) = o1;
}

// =====================================================================
// Kernel 3: projection GEMM  out[M,128] = attn[M,128] @ proj_w^T + proj_b
// Both A and B tiles staged with async global->LDS b128 copies.
// =====================================================================
__global__ __launch_bounds__(256)
void proj_gemm_kernel(const _Float16* __restrict__ aws,
                      const _Float16* __restrict__ wproj,
                      const float* __restrict__ pb,
                      float* __restrict__ out)
{
    __shared__ _Float16 As[64 * 40];
    __shared__ _Float16 Bs[64 * 40];

    const int tid  = threadIdx.x;
    const int lane = tid & 31;
    const int wave = tid >> 5;
    const int rowBase = blockIdx.x * 64;
    const int colBase = blockIdx.y * 64;
    const int subM  = wave >> 1;
    const int subN0 = (wave & 1) * 2;

    const int br = tid >> 2, bc = (tid & 3) * 8;     // one 16B chunk per thread

    v8f acc0 = {};
    v8f acc1 = {};

    for (int kk = 0; kk < 4; ++kk) {
        const int k0 = kk * 32;
        async_copy_b128(aws + (size_t)(rowBase + br) * CC + k0 + bc,
                        &As[br * 40 + bc]);
        async_copy_b128(wproj + (size_t)(colBase + br) * CC + k0 + bc,
                        &Bs[br * 40 + bc]);
        wait_async0();
        __syncthreads();

        const int mrow = subM * 16 + (lane & 15);
        const int ka   = ((lane >> 4) & 1) * 8;
        v16h a = cat16(*(const v8h*)&As[mrow * 40 + ka],
                       *(const v8h*)&As[mrow * 40 + ka + 16]);

        const int kb    = ((lane >> 4) & 1) * 8;
        const int nrow0 = subN0 * 16 + (lane & 15);
        v16h bf0 = cat16(*(const v8h*)&Bs[nrow0 * 40 + kb],
                         *(const v8h*)&Bs[nrow0 * 40 + kb + 16]);
        v16h bf1 = cat16(*(const v8h*)&Bs[(nrow0 + 16) * 40 + kb],
                         *(const v8h*)&Bs[(nrow0 + 16) * 40 + kb + 16]);

        acc0 = __builtin_amdgcn_wmma_f32_16x16x32_f16(false, a, false, bf0,
                                                      (short)0, acc0, false, false);
        acc1 = __builtin_amdgcn_wmma_f32_16x16x32_f16(false, a, false, bf1,
                                                      (short)0, acc1, false, false);
        __syncthreads();
    }

    const int n  = lane & 15;
    const int hs = (lane >> 4) & 1;
    union { v8f v; float f[8]; } u0, u1;
    u0.v = acc0; u1.v = acc1;
    #pragma unroll
    for (int t = 0; t < 2; ++t) {
        const float* af = t ? u1.f : u0.f;
        const int gc = colBase + (subN0 + t) * 16 + n;
        const float bias = pb[gc];
        #pragma unroll
        for (int r = 0; r < 8; ++r) {
            const int gm = rowBase + subM * 16 + hs * 8 + r;
            out[(size_t)gm * CC + gc] = af[r] + bias;
        }
    }
}

// =====================================================================
extern "C" void kernel_launch(void* const* d_in, const int* in_sizes, int n_in,
                              void* d_out, int out_size, void* d_ws, size_t ws_size,
                              hipStream_t stream) {
    (void)in_sizes; (void)n_in; (void)out_size; (void)ws_size;
    const float* x      = (const float*)d_in[0];
    const float* q_w    = (const float*)d_in[1];
    const float* q_b    = (const float*)d_in[2];
    const float* kv_w   = (const float*)d_in[3];
    const float* kv_b   = (const float*)d_in[4];
    const float* proj_w = (const float*)d_in[5];
    const float* proj_b = (const float*)d_in[6];
    const float* temp   = (const float*)d_in[7];
    const float* qe     = (const float*)d_in[8];   // (8,1,16) flattened = 128
    const float* rpb    = (const float*)d_in[9];   // (8,27)

    _Float16* ws = (_Float16*)d_ws;
    const size_t perBuf = (size_t)MTOT * CC;       // f16 elements per buffer
    _Float16* qws  = ws;
    _Float16* kws  = ws + perBuf;
    _Float16* vws  = ws + 2 * perBuf;
    _Float16* ows  = ws + 3 * perBuf;
    _Float16* wf16 = ws + 4 * perBuf;              // fused f16 weights [512,128]
    _Float16* wqkv = wf16;                         // rows 0..383
    _Float16* wprj = wf16 + (size_t)384 * CC;      // rows 384..511

    dim3 blk(256);
    wconv_kernel<<<dim3((512 * CC) / (256 * 4)), blk, 0, stream>>>(
        q_w, kv_w, proj_w, wf16);
    qkv_gemm_kernel<<<dim3(MTOT / 64, 384 / 64), blk, 0, stream>>>(
        x, wqkv, q_b, kv_b, qe, qws, kws, vws);
    swin_attn_kernel<<<dim3((BB * NHEADS * NN) / 256), blk, 0, stream>>>(
        qws, kws, vws, ows, temp, rpb);
    proj_gemm_kernel<<<dim3(MTOT / 64, CC / 64), blk, 0, stream>>>(
        ows, wprj, proj_b, (float*)d_out);
}
